// AttentionalGNN_62139586839238
// MI455X (gfx1250) — matrix operand order repaired and checked by reading, not compiled, not run on hardware
//
#include <hip/hip_runtime.h>
#include <hip/hip_bf16.h>

typedef __attribute__((ext_vector_type(16))) _Float16 v16h;
typedef __attribute__((ext_vector_type(8)))  float    v8f;

union Frag {
  v16h v;
  _Float16 h[16];
  float4 q[2];
};

#define MODE_NONE  0
#define MODE_ELU1  1   // elu(x)+1  (q,k pre-activation)
#define MODE_SCALE 2   // (x)*scale (v / m)
#define MODE_ELU   3   // elu(x)    (GAT output)

// ---------------------------------------------------------------------------
// Tiled WMMA GEMM:  Out[b,o,n] = epilogue( sum_c A[o,c] * X[b,c,n] + bias[o] )
// A is MxK row-major (or KxM when TRANS => effective A[o,c] = A[c*M+o]).
// If SPLIT, X channels 0..Ksplit-1 come from X0, the rest from X1 ([x;msg]).
// Block tile: 128(M) x 128(N), K-step 32. 8 waves: wave w owns rows
// [16w,16w+16) and all 128 cols => 1 A frag reused across 8 B frags,
// 8 WMMAs per wave per K-step. Global loads for step k+1 are issued before
// the WMMA block of step k (register prefetch pipeline).
// ---------------------------------------------------------------------------
template <bool TRANS, bool SPLIT>
__global__ __launch_bounds__(256) void gemm_conv(
    const float* __restrict__ A, const float* __restrict__ X0,
    const float* __restrict__ X1, const float* __restrict__ bias,
    const float* __restrict__ Res, float* __restrict__ Out,
    int M, int K, int N, int Ksplit, int mode, float scale)
{
  __shared__ __align__(16) _Float16 lA[128][40];
  __shared__ __align__(16) _Float16 lX[128][40];

  const int tid  = threadIdx.x;
  const int wave = tid >> 5;
  const int lane = tid & 31;
  const int lm   = lane & 15;
  const int hi   = lane >> 4;
  const int m0 = blockIdx.y * 128, n0 = blockIdx.x * 128, b = blockIdx.z;
  const int K1 = K - Ksplit;

  // staging assignments
  const int arow = tid >> 1;          // non-trans A: 128 rows x 32 k, 16 halves/thread
  const int acol = (tid & 1) << 4;
  const int tkk  = (tid >> 4) << 1;   // k-row pair (0,2,..,30) for X and trans-A
  const int tcc  = (tid & 15) << 3;   // 8 columns (of 128)

  float aa[16];          // A tile regs (16 floats)
  float x0[8], x1[8];    // X tile regs: rows tkk, tkk+1

  auto loadTile = [&](int k0) {
    if (!TRANS) {
      const float* ap = A + (size_t)(m0 + arow) * K + (k0 + acol);
      ((float4*)aa)[0] = ((const float4*)ap)[0];
      ((float4*)aa)[1] = ((const float4*)ap)[1];
      ((float4*)aa)[2] = ((const float4*)ap)[2];
      ((float4*)aa)[3] = ((const float4*)ap)[3];
    } else {
      const float* ap0 = A + (size_t)(k0 + tkk) * M + (m0 + tcc);
      const float* ap1 = ap0 + M;
      ((float4*)aa)[0] = ((const float4*)ap0)[0];
      ((float4*)aa)[1] = ((const float4*)ap0)[1];
      ((float4*)aa)[2] = ((const float4*)ap1)[0];
      ((float4*)aa)[3] = ((const float4*)ap1)[1];
    }
    int cg = k0 + tkk;
    const float* xp0;
    const float* xp1;
    if (SPLIT) {
      if (cg < Ksplit)     xp0 = X0 + ((size_t)b * Ksplit + cg) * N + n0 + tcc;
      else                 xp0 = X1 + ((size_t)b * K1 + (cg - Ksplit)) * N + n0 + tcc;
      if (cg + 1 < Ksplit) xp1 = X0 + ((size_t)b * Ksplit + cg + 1) * N + n0 + tcc;
      else                 xp1 = X1 + ((size_t)b * K1 + (cg + 1 - Ksplit)) * N + n0 + tcc;
    } else {
      xp0 = X0 + ((size_t)b * K + cg) * N + n0 + tcc;
      xp1 = xp0 + N;
    }
    ((float4*)x0)[0] = ((const float4*)xp0)[0];
    ((float4*)x0)[1] = ((const float4*)xp0)[1];
    ((float4*)x1)[0] = ((const float4*)xp1)[0];
    ((float4*)x1)[1] = ((const float4*)xp1)[1];
  };

  auto storeTile = [&]() {
    if (!TRANS) {
      union { _Float16 h[16]; uint4 q[2]; } p;
#pragma unroll
      for (int i = 0; i < 16; ++i) p.h[i] = (_Float16)aa[i];
      *(uint4*)&lA[arow][acol]     = p.q[0];   // ds_store_b128
      *(uint4*)&lA[arow][acol + 8] = p.q[1];
    } else {
#pragma unroll
      for (int i = 0; i < 8; ++i) {           // pack (k,k+1) pair -> one b32 store
        union { _Float16 h[2]; unsigned u; } p;
        p.h[0] = (_Float16)aa[i];
        p.h[1] = (_Float16)aa[8 + i];
        *(unsigned*)&lA[tcc + i][tkk] = p.u;
      }
    }
#pragma unroll
    for (int i = 0; i < 8; ++i) {             // X transposed: pack k-pair -> b32
      union { _Float16 h[2]; unsigned u; } p;
      p.h[0] = (_Float16)x0[i];
      p.h[1] = (_Float16)x1[i];
      *(unsigned*)&lX[tcc + i][tkk] = p.u;
    }
  };

  v8f acc[8] = {};

  loadTile(0);
  for (int k0 = 0; k0 < K; k0 += 32) {
    storeTile();
    __syncthreads();
    if (k0 + 32 < K) loadTile(k0 + 32);  // overlap next global loads with WMMAs

    // A 16x32 f16 frag: lane(lm,hi): row=lm; halves j<8 -> K=hi*8+j,
    // j>=8 -> K=16+hi*8+(j-8): two contiguous 16B chunks.
    Frag af;
    {
      const _Float16* ap0 = &lA[wave * 16 + lm][hi * 8];
      af.q[0] = *(const float4*)ap0;
      af.q[1] = *(const float4*)(ap0 + 16);
    }
#pragma unroll
    for (int t = 0; t < 8; ++t) {
      // B 32x16 f16 frag: N = lane&15, K same as A: hi*8+j (j<8), 16+hi*8+(j-8) (j>=8).
      Frag bf;
      const _Float16* bp = &lX[t * 16 + lm][hi * 8];
      bf.q[0] = *(const float4*)bp;
      bf.q[1] = *(const float4*)(bp + 16);
      acc[t] = __builtin_amdgcn_wmma_f32_16x16x32_f16(
          false, af.v, false, bf.v, (short)0, acc[t], false, false);
    }
    __syncthreads();
  }

  // ---- epilogue: C/D layout row = r + 8*hi, col = lm ----
#pragma unroll
  for (int t = 0; t < 8; ++t) {
    int col = n0 + t * 16 + lm;
#pragma unroll
    for (int r = 0; r < 8; ++r) {
      int row = m0 + wave * 16 + hi * 8 + r;
      float v = acc[t][r];
      if (bias) v += bias[row];
      if (mode == MODE_ELU1)       v = v > 0.f ? v + 1.f : __expf(v);   // elu+1
      else if (mode == MODE_SCALE) v *= scale;
      else if (mode == MODE_ELU)   v = v > 0.f ? v : __expf(v) - 1.f;
      size_t idx = ((size_t)b * M + row) * N + col;
      if (Res) v += Res[idx];
      Out[idx] = v;
    }
  }
}

// ---------------------------------------------------------------------------
// Per-row sum over N: out[row] = sum_n X[row*N+n]   (row = b*256 + c)
// ---------------------------------------------------------------------------
__global__ __launch_bounds__(256) void rowsum_kernel(
    const float* __restrict__ X, float* __restrict__ out, int N)
{
  size_t row = blockIdx.x;
  const float* p = X + row * (size_t)N;
  float s = 0.f;
  for (int i = threadIdx.x; i < N; i += 256) s += p[i];
  for (int o = 16; o > 0; o >>= 1) s += __shfl_down(s, o);
  __shared__ float sh[8];
  if ((threadIdx.x & 31) == 0) sh[threadIdx.x >> 5] = s;
  __syncthreads();
  if (threadIdx.x == 0) {
    float t = 0.f;
    for (int w = 0; w < 8; ++w) t += sh[w];
    out[row] = t;
  }
}

// ---------------------------------------------------------------------------
// In-place instance-norm over N per (b,channel) row, then ReLU.
// ---------------------------------------------------------------------------
__global__ __launch_bounds__(256) void instnorm_relu_kernel(float* __restrict__ X, int N)
{
  size_t row = blockIdx.x;
  float* p = X + row * (size_t)N;
  float s = 0.f, ss = 0.f;
  for (int i = threadIdx.x; i < N; i += 256) { float v = p[i]; s += v; ss += v * v; }
  for (int o = 16; o > 0; o >>= 1) { s += __shfl_down(s, o); ss += __shfl_down(ss, o); }
  __shared__ float shs[8], shq[8], mean_s, rstd_s;
  if ((threadIdx.x & 31) == 0) { shs[threadIdx.x >> 5] = s; shq[threadIdx.x >> 5] = ss; }
  __syncthreads();
  if (threadIdx.x == 0) {
    float ts = 0.f, tq = 0.f;
    for (int w = 0; w < 8; ++w) { ts += shs[w]; tq += shq[w]; }
    float mean = ts / N;
    float var  = tq / N - mean * mean;
    mean_s = mean;
    rstd_s = rsqrtf(var + 1e-5f);
  }
  __syncthreads();
  float mean = mean_s, rstd = rstd_s;
  for (int i = threadIdx.x; i < N; i += 256) {
    float v = (p[i] - mean) * rstd;
    p[i] = v > 0.f ? v : 0.f;
  }
}

// ---------------------------------------------------------------------------
// KV[b,h,q,d] = sum_m k[b,d,h,m] * v[b,q,h,m]   (channel c = d*4 + h; v pre-scaled 1/m)
// One block per (b,h); 64x64 output; m staged in LDS chunks of 64.
// ---------------------------------------------------------------------------
__global__ __launch_bounds__(256) void kv_kernel(
    const float* __restrict__ Kb, const float* __restrict__ Vb,
    float* __restrict__ KV, int Nsrc)
{
  const int b = blockIdx.x >> 2, h = blockIdx.x & 3;
  __shared__ float kc[64][65];
  __shared__ float vc[64][65];
  const int tid = threadIdx.x;
  const int qd = tid & 63, dg = (tid >> 6) << 4;
  float acc[16];
#pragma unroll
  for (int j = 0; j < 16; ++j) acc[j] = 0.f;

  for (int m0 = 0; m0 < Nsrc; m0 += 64) {
#pragma unroll
    for (int i = 0; i < 16; ++i) {
      int idx = tid + (i << 8);
      int row = idx >> 6, mm = idx & 63;
      size_t off = ((size_t)b * 256 + row * 4 + h) * (size_t)Nsrc + m0 + mm;
      kc[row][mm] = Kb[off];
      vc[row][mm] = Vb[off];
    }
    __syncthreads();
    for (int mm = 0; mm < 64; ++mm) {
      float vv = vc[qd][mm];
#pragma unroll
      for (int j = 0; j < 16; ++j) acc[j] += vv * kc[dg + j][mm];
    }
    __syncthreads();
  }
  size_t base = ((size_t)blockIdx.x * 64 + qd) * 64 + dg;
#pragma unroll
  for (int j = 0; j < 16; ++j) KV[base + j] = acc[j];
}

// ---------------------------------------------------------------------------
// out[b, qd*4+h, m] = m_src * Z * sum_d q[b,d,h,m] * KV[b,h,qd,d]
// Z = 1/(sum_d q*ksum + eps). One block per (b,h, 64-m chunk).
// ---------------------------------------------------------------------------
__global__ __launch_bounds__(256) void attn_out_kernel(
    const float* __restrict__ Q, const float* __restrict__ KV,
    const float* __restrict__ ksum, float* __restrict__ Out, int Nx, float scale)
{
  const int bh = blockIdx.y, b = bh >> 2, h = bh & 3;
  const int m0 = blockIdx.x * 64;
  __shared__ float kvs[64][65];
  __shared__ float qs[64][65];
  __shared__ float zs[64];
  __shared__ float ksv[64];
  const int tid = threadIdx.x;
#pragma unroll
  for (int i = 0; i < 16; ++i) {
    int idx = tid + (i << 8);
    int row = idx >> 6, cc = idx & 63;
    kvs[row][cc] = KV[(size_t)bh * 4096 + idx];
    qs[row][cc]  = Q[((size_t)b * 256 + row * 4 + h) * (size_t)Nx + m0 + cc];
  }
  if (tid < 64) ksv[tid] = ksum[(size_t)b * 256 + tid * 4 + h];
  __syncthreads();
  if (tid < 64) {
    float s = 0.f;
    for (int d = 0; d < 64; ++d) s += qs[d][tid] * ksv[d];
    zs[tid] = scale / (s + 1e-6f);
  }
  __syncthreads();
  const int qd = tid & 63, mg = (tid >> 6) << 4;
  for (int i = 0; i < 16; ++i) {
    int mm = mg + i;
    float s = 0.f;
    for (int d = 0; d < 64; ++d) s += kvs[qd][d] * qs[d][mm];
    Out[((size_t)b * 256 + qd * 4 + h) * (size_t)Nx + m0 + mm] = s * zs[mm];
  }
}

// ---------------------------------------------------------------------------
// GAT fused scores+softmax+aggregate:
// e_l = leaky_relu(d3.(W a1) + d2_l.(W a2)); attn = softmax_l(e);
// xagg[b,d,n] = attn_0*d3 + sum_l attn_l*d2_l. One block per (b,n), thread=d.
// ---------------------------------------------------------------------------
__global__ __launch_bounds__(256) void gat_kernel(
    const float* __restrict__ D3, const float* __restrict__ D2,
    const float* __restrict__ wa, float* __restrict__ XA, int N3)
{
  const int n = blockIdx.x, b = blockIdx.y;
  const int d = threadIdx.x;
  const int lane = d & 31, wave = d >> 5;
  float d3v = D3[((size_t)b * 256 + d) * (size_t)N3 + n];
  const float* p2 = D2 + ((size_t)b * 256 + d) * ((size_t)N3 * 8) + (size_t)n * 8;
  float4 la = ((const float4*)p2)[0];
  float4 lb = ((const float4*)p2)[1];
  float l2[8] = { la.x, la.y, la.z, la.w, lb.x, lb.y, lb.z, lb.w };
  float wa1v = wa[d], wa2v = wa[256 + d];
  float vals[10];
  vals[0] = d3v * wa1v;
  vals[1] = d3v * wa2v;
#pragma unroll
  for (int i = 0; i < 8; ++i) vals[2 + i] = l2[i] * wa2v;
#pragma unroll
  for (int i = 0; i < 10; ++i)
    for (int o = 16; o > 0; o >>= 1) vals[i] += __shfl_down(vals[i], o);
  __shared__ float sh[8][10];
  __shared__ float attn[9];
  if (lane == 0) {
#pragma unroll
    for (int i = 0; i < 10; ++i) sh[wave][i] = vals[i];
  }
  __syncthreads();
  if (d == 0) {
    float t[10];
#pragma unroll
    for (int i = 0; i < 10; ++i) {
      float s = 0.f;
      for (int w = 0; w < 8; ++w) s += sh[w][i];
      t[i] = s;
    }
    float e[9], mx = -1e30f;
#pragma unroll
    for (int l = 0; l < 9; ++l) {
      float x = t[0] + t[1 + l];
      x = x > 0.f ? x : 0.2f * x;     // leaky_relu alpha=0.2
      e[l] = x;
      mx = fmaxf(mx, x);
    }
    float ssum = 0.f;
#pragma unroll
    for (int l = 0; l < 9; ++l) { float ex = __expf(e[l] - mx); attn[l] = ex; ssum += ex; }
    float inv = 1.0f / ssum;
#pragma unroll
    for (int l = 0; l < 9; ++l) attn[l] *= inv;
  }
  __syncthreads();
  float r = attn[0] * d3v;
#pragma unroll
  for (int i = 0; i < 8; ++i) r += attn[1 + i] * l2[i];
  XA[((size_t)b * 256 + d) * (size_t)N3 + n] = r;
}

// wa[d] = sum_e W[d,e]*a1[e]; wa[256+d] = sum_e W[d,e]*a2[e]
__global__ __launch_bounds__(256) void wa_kernel(
    const float* __restrict__ W, const float* __restrict__ a, float* __restrict__ wa)
{
  int d = threadIdx.x;
  const float* row = W + (size_t)d * 256;
  float s1 = 0.f, s2 = 0.f;
  for (int e = 0; e < 256; ++e) {
    float w = row[e];
    s1 += w * a[e];
    s2 += w * a[256 + e];
  }
  wa[d] = s1;
  wa[256 + d] = s2;
}

__global__ __launch_bounds__(256) void add_kernel(
    float* __restrict__ dst, const float* __restrict__ src, int n4)
{
  int i = blockIdx.x * 256 + threadIdx.x;
  if (i < n4) {
    float4 a = ((const float4*)src)[i];
    float4 d = ((float4*)dst)[i];
    d.x += a.x; d.y += a.y; d.z += a.z; d.w += a.w;
    ((float4*)dst)[i] = d;
  }
}

// ---------------------------------------------------------------------------
extern "C" void kernel_launch(void* const* d_in, const int* in_sizes, int n_in,
                              void* d_out, int out_size, void* d_ws, size_t ws_size,
                              hipStream_t stream)
{
  (void)in_sizes; (void)n_in; (void)out_size; (void)ws_size;
  const int B = 8, D = 256, NQ = 4096, N3 = 2048;
  const size_t DD  = (size_t)D * D;           // 65536
  const size_t SZQ = (size_t)B * D * NQ;      // 8388608
  const size_t SZ3 = (size_t)B * D * N3;      // 4194304

  float* out2d = (float*)d_out;               // (B,D,NQ) live buffer == output 0
  float* out3d = out2d + SZQ;                 // (B,D,N3) live buffer == output 1
  hipMemcpyAsync(out2d, d_in[0], SZQ * sizeof(float), hipMemcpyDeviceToDevice, stream);
  hipMemcpyAsync(out3d, d_in[1], SZ3 * sizeof(float), hipMemcpyDeviceToDevice, stream);

  const float* d2db = (const float*)d_in[2];
  const float* gatW = (const float*)d_in[3];
  const float* gata = (const float*)d_in[4];
  const float* pw   = (const float*)d_in[5];
  const float* pb   = (const float*)d_in[6];
  const float* mw   = (const float*)d_in[7];
  const float* mb   = (const float*)d_in[8];
  const float* m1w  = (const float*)d_in[9];
  const float* m1b  = (const float*)d_in[10];
  const float* m2w  = (const float*)d_in[11];
  const float* m2b  = (const float*)d_in[12];

  float* ws = (float*)d_ws;
  size_t o = 0;
  float* W1  = ws + o; o += SZQ;   // q / xagg
  float* W2  = ws + o; o += SZQ;   // k
  float* W3  = ws + o; o += SZQ;   // v
  float* W4  = ws + o; o += SZQ;   // att
  float* W5  = ws + o; o += SZQ;   // msg
  float* DL  = ws + o; o += SZQ;   // delta0
  float* T   = ws + o; o += 2 * SZQ; // m1 output (512 channels)
  float* KVb = ws + o; o += (size_t)B * 4 * 64 * 64;
  float* KS  = ws + o; o += (size_t)B * 256;
  float* WA  = ws + o; o += 512;

  // plain GEMM (A row-major, single X source)
  auto gemm = [&](const float* A, const float* X0, const float* bias,
                  const float* Res, float* Out,
                  int M, int K, int N, int mode, float scale) {
    dim3 g(N / 128, M / 128, B);
    gemm_conv<false, false><<<g, 256, 0, stream>>>(A, X0, X0, bias, Res, Out,
                                                   M, K, N, K, mode, scale);
  };
  // split-K-source GEMM (X = [X0(Ksplit ch) ; X1(K-Ksplit ch)])
  auto gemm_cat = [&](const float* A, const float* X0, const float* X1,
                      const float* bias, float* Out,
                      int M, int K, int N, int Ksplit) {
    dim3 g(N / 128, M / 128, B);
    gemm_conv<false, true><<<g, 256, 0, stream>>>(A, X0, X1, bias, nullptr, Out,
                                                  M, K, N, Ksplit, MODE_NONE, 0.f);
  };
  // transposed-A GEMM (effective A[o,c] = A[c*M+o]); GAT output
  auto gemm_tr = [&](const float* A, const float* X0, float* Out,
                     int M, int K, int N, int mode) {
    dim3 g(N / 128, M / 128, B);
    gemm_conv<true, false><<<g, 256, 0, stream>>>(A, X0, X0, nullptr, nullptr, Out,
                                                  M, K, N, K, mode, 0.f);
  };

  // attn_prop(x, src) -> Out (= x + delta if Res, else delta)
  auto attn = [&](const float* x, int Nx, const float* src, int Nsrc,
                  float* outp, const float* res, int ai) {
    const float* PW = pw + (size_t)ai * 3 * DD;
    const float* PB = pb + (size_t)ai * 3 * D;
    gemm(PW,        x,   PB,       nullptr, W1, D, D, Nx,   MODE_ELU1, 0.f);          // q
    gemm(PW + DD,   src, PB + D,   nullptr, W2, D, D, Nsrc, MODE_ELU1, 0.f);          // k
    gemm(PW + 2*DD, src, PB + 2*D, nullptr, W3, D, D, Nsrc, MODE_SCALE, 1.0f / Nsrc); // v/m
    rowsum_kernel<<<B * D, 256, 0, stream>>>(W2, KS, Nsrc);
    kv_kernel<<<B * 4, 256, 0, stream>>>(W2, W3, KVb, Nsrc);
    attn_out_kernel<<<dim3(Nx / 64, B * 4), 256, 0, stream>>>(W1, KVb, KS, W4, Nx, (float)Nsrc);
    gemm(mw + (size_t)ai * DD, W4, mb + (size_t)ai * D, nullptr, W5, D, D, Nx, MODE_NONE, 0.f); // msg
    gemm_cat(m1w + (size_t)ai * 4 * DD, x, W5, m1b + (size_t)ai * 2 * D, T,
             2 * D, 2 * D, Nx, D);                                                    // m1 on [x;msg]
    instnorm_relu_kernel<<<B * 2 * D, 256, 0, stream>>>(T, Nx);
    gemm(m2w + (size_t)ai * 2 * DD, T, m2b + (size_t)ai * D, res, outp,
         D, 2 * D, Nx, MODE_NONE, 0.f);                                               // m2 (+res)
  };

  int gi = 0, ai = 0;
  for (int li = 0; li < 9; ++li) {
    int kind = li % 3; // 0=GATs, 1=self, 2=cross
    if (kind == 0) {
      wa_kernel<<<1, 256, 0, stream>>>(gatW + (size_t)gi * DD, gata + (size_t)gi * 512, WA);
      gat_kernel<<<dim3(N3, B), 256, 0, stream>>>(out3d, d2db, WA, W1, N3);
      // out3d = elu( W^T . xagg )
      gemm_tr(gatW + (size_t)gi * DD, W1, out3d, D, D, N3, MODE_ELU);
      ++gi;
    } else {
      bool cross = (kind == 2);
      const float* s0 = cross ? out3d : out2d; int N0 = cross ? N3 : NQ;
      const float* s1 = cross ? out2d : out3d; int N1 = cross ? NQ : N3;
      // delta0 for desc2d_query goes to DL (must not update before pipeline 1 reads old tensors)
      attn(out2d, NQ, s0, N0, DL,    nullptr, ai);
      // desc3d update in place (its own reads complete before the residual write)
      attn(out3d, N3, s1, N1, out3d, out3d,  ai);
      add_kernel<<<(int)(SZQ / 4 / 256), 256, 0, stream>>>(out2d, DL, (int)(SZQ / 4));
      ++ai;
    }
  }
}
